// TGN_45363444580436
// MI455X (gfx1250) — matrix operand order repaired and checked by reading, not compiled, not run on hardware
//
#include <hip/hip_runtime.h>

// ---- problem constants (TGN Wikipedia-style) ----
#define KN    20
#define FDIM  172
#define TDIM  172
#define DDIM  204
#define DH    102       // DDIM / 2 heads
#define KV_F  516       // F + F + T
#define KV_LD 544       // 516 padded to 17*32
#define A1_LD 192       // 172 padded to 32-mult
#define Q_LD  232       // 204 + head-split pad, 16B-aligned head offsets (0,104)
#define MRG_LD 384      // 376 padded
#define HID_LD 224      // 204 padded
#define ATT_LD 208

typedef __attribute__((ext_vector_type(16))) _Float16 v16h;
typedef __attribute__((ext_vector_type(8)))  _Float16 v8h;
typedef __attribute__((ext_vector_type(8)))  float    v8f;

union V16U { v16h v; v8h h[2]; };

// A fragment (16x32 f16): lane L<16 holds row L, K {kb..kb+7} and {kb+16..kb+23}, kb=(L>>4)*8
__device__ __forceinline__ v16h frag_a(const _Float16* A, int lda, int row0, int k0, int lane) {
  int r  = row0 + (lane & 15);
  int kb = k0 + ((lane >> 4) << 3);
  const _Float16* p = A + (size_t)r * lda + kb;
  V16U u;
  u.h[0] = *(const v8h*)(p);
  u.h[1] = *(const v8h*)(p + 16);
  return u.v;
}
// B fragment (32x16 f16) from BT[N][K]: lane L holds col (L&15), K {kb..kb+7} and {kb+16..kb+23}, kb=(L>>4)*8
__device__ __forceinline__ v16h frag_b(const _Float16* BT, int ldbt, int n0, int k0, int lane) {
  int n  = n0 + (lane & 15);
  int kb = k0 + ((lane >> 4) << 3);
  const _Float16* p = BT + (size_t)n * ldbt + kb;
  V16U u;
  u.h[0] = *(const v8h*)(p);
  u.h[1] = *(const v8h*)(p + 16);
  return u.v;
}

// ---------------- generic WMMA GEMM: C = act(A @ BT^T + bias) ----------------
// A: [M, lda] f16 row-major. BT: [nTiles*16, ldbt] f16, zero-padded past real N and K.
// One 16x16 output tile per wave; 4 waves / block; blockIdx.x tiles M, blockIdx.y*4+wave tiles N.
__global__ void wmma_gemm(const _Float16* __restrict__ A, int lda, int kpad,
                          const _Float16* __restrict__ BT, int ldbt, int nTiles,
                          const float* __restrict__ bias,
                          _Float16* __restrict__ C16, float* __restrict__ C32,
                          int ldc, int nreal, int outCol0, int relu, int remap)
{
  const int lane  = threadIdx.x & 31;
  const int wave  = threadIdx.x >> 5;
  const int ntile = blockIdx.y * 4 + wave;
  if (ntile >= nTiles) return;                 // wave-uniform: EXEC stays all-ones
  const int row0 = blockIdx.x << 4;
  const int n0   = ntile << 4;

  v8f acc = {};
  for (int k0 = 0; k0 < kpad; k0 += 32) {
    v16h a = frag_a(A, lda, row0, k0, lane);
    v16h b = frag_b(BT, ldbt, n0, k0, lane);
    acc = __builtin_amdgcn_wmma_f32_16x16x32_f16(false, a, false, b, (short)0, acc,
                                                 false, false);
  }

  const int n = n0 + (lane & 15);
  if (n < nreal) {
    const int c     = outCol0 + n + ((remap && n >= DH) ? 2 : 0);
    const float bv  = bias ? bias[n] : 0.0f;
    const int rbase = row0 + ((lane >> 4) << 3);   // C/D layout: vgpr r -> row r (+8 for hi lanes)
    #pragma unroll
    for (int r = 0; r < 8; ++r) {
      float v = acc[r] + bv;
      if (relu) v = fmaxf(v, 0.0f);
      if (C32) C32[(size_t)(rbase + r) * ldc + c] = v;
      else     C16[(size_t)(rbase + r) * ldc + c] = (_Float16)v;
    }
  }
}

// ---------------- weight packing: transposed  dst[n][k] = src[k0+k][n0+n] ----------------
__global__ void pack_t(_Float16* __restrict__ dst, int dld, const float* __restrict__ src,
                       int sld, int n0, int k0, int nr, int kr, int tot) {
  int i = blockIdx.x * 256 + threadIdx.x;
  if (i >= tot) return;
  int n = i / dld, k = i % dld;
  float v = (n < nr && k < kr) ? src[(size_t)(k0 + k) * sld + (n0 + n)] : 0.0f;
  dst[i] = (_Float16)v;
}
// direct  dst[n][k] = src[r0+n][c0+k]
__global__ void pack_d(_Float16* __restrict__ dst, int dld, const float* __restrict__ src,
                       int sld, int r0, int c0, int rr, int cr, int tot) {
  int i = blockIdx.x * 256 + threadIdx.x;
  if (i >= tot) return;
  int n = i / dld, k = i % dld;
  float v = (n < rr && k < cr) ? src[(size_t)(r0 + n) * sld + (c0 + k)] : 0.0f;
  dst[i] = (_Float16)v;
}

// qbias[n] = sum_t cos(time_b[t]) * Wq[172+t][n]   (phi0 is query-independent)
__global__ void qbias_kernel(float* __restrict__ qb, const float* __restrict__ tb,
                             const float* __restrict__ Wq) {
  int n = threadIdx.x;
  if (n >= DDIM) return;
  float s = 0.0f;
  for (int t = 0; t < TDIM; ++t)
    s += cosf(tb[t]) * Wq[(size_t)(FDIM + t) * DDIM + n];
  qb[n] = s;
}

// A1[q][k] = f16(node_features[nodes[q]][k] + memory[nodes[q]][k])
__global__ void build_a1(_Float16* __restrict__ A1, const float* __restrict__ nf,
                         const float* __restrict__ mem, const int* __restrict__ nodes, int tot) {
  int i = blockIdx.x * 256 + threadIdx.x;
  if (i >= tot) return;
  int q = i / FDIM, k = i % FDIM;
  int n = nodes[q];
  A1[(size_t)q * A1_LD + k] = (_Float16)(nf[(size_t)n * FDIM + k] + mem[(size_t)n * FDIM + k]);
}

// merged[q][c] = c<204 ? att_out[q][c] : node_features[nodes[q]][c-204]
__global__ void build_merged(_Float16* __restrict__ M, const _Float16* __restrict__ att,
                             const float* __restrict__ nf, const int* __restrict__ nodes, int tot) {
  int i = blockIdx.x * 256 + threadIdx.x;
  if (i >= tot) return;
  int q = i / 376, c = i % 376;
  float v = (c < DDIM) ? (float)att[(size_t)q * ATT_LD + c]
                       : nf[(size_t)nodes[q] * FDIM + (c - DDIM)];
  M[(size_t)q * MRG_LD + c] = (_Float16)v;
}

// ---------------- fused gather + time-encode + scores + softmax + ctx ----------------
// one block (256 threads) per query
__global__ void attn_kernel(const float* __restrict__ nf, const float* __restrict__ ef,
                            const float* __restrict__ mem, const float* __restrict__ tw,
                            const float* __restrict__ tb, const float* __restrict__ ts,
                            const float* __restrict__ ntm, const int* __restrict__ nghs,
                            const int* __restrict__ eidx,
                            const _Float16* __restrict__ G0, const _Float16* __restrict__ G1,
                            _Float16* __restrict__ ctx0, _Float16* __restrict__ ctx1)
{
  __shared__ _Float16 kv[KN][KV_LD];
  __shared__ _Float16 g[2][KV_LD];
  __shared__ float sc[2][KN];
  __shared__ float at[2][KN];
  __shared__ int   sng[KN];
  __shared__ int   sei[KN];
  __shared__ float snt[KN];

  const int q = blockIdx.x, tid = threadIdx.x;
  if (tid < KN) {
    sng[tid] = nghs[q * KN + tid];
    sei[tid] = eidx[q * KN + tid];
    snt[tid] = ntm[q * KN + tid];
  }
  __syncthreads();
  const float tq = ts[q];

  // phase 1: build kv rows (f16) + stage G rows. quads of 4 floats; segment bounds 172/344/516 are 4-mult
  for (int idx = tid; idx < KN * (KV_LD / 4); idx += 256) {
    int k = idx / (KV_LD / 4);
    int f = (idx % (KV_LD / 4)) * 4;
    float x0, x1, x2, x3;
    if (f < FDIM) {
      int n = sng[k];
      float4 a = *(const float4*)(nf + (size_t)n * FDIM + f);
      float4 b = *(const float4*)(mem + (size_t)n * FDIM + f);
      x0 = a.x + b.x; x1 = a.y + b.y; x2 = a.z + b.z; x3 = a.w + b.w;
    } else if (f < 2 * FDIM) {
      float4 e = *(const float4*)(ef + (size_t)sei[k] * FDIM + (f - FDIM));
      x0 = e.x; x1 = e.y; x2 = e.z; x3 = e.w;
    } else if (f < KV_F) {
      int t = f - 2 * FDIM;
      float dt = tq - snt[k];
      float4 w = *(const float4*)(tw + t);
      float4 b = *(const float4*)(tb + t);
      x0 = cosf(dt * w.x + b.x); x1 = cosf(dt * w.y + b.y);
      x2 = cosf(dt * w.z + b.z); x3 = cosf(dt * w.w + b.w);
    } else {
      x0 = x1 = x2 = x3 = 0.0f;
    }
    _Float16* p = &kv[k][f];
    p[0] = (_Float16)x0; p[1] = (_Float16)x1; p[2] = (_Float16)x2; p[3] = (_Float16)x3;
  }
  for (int idx = tid; idx < 2 * (KV_LD / 4); idx += 256) {
    int h = idx / (KV_LD / 4);
    int f = (idx % (KV_LD / 4)) * 4;
    const _Float16* gp = (h ? G1 : G0) + (size_t)q * KV_LD + f;
    *(uint2*)&g[h][f] = *(const uint2*)gp;
  }
  __syncthreads();

  // phase 2: scores[h][k] = (kv[k] . G[h]) / sqrt(dh).  8 waves over 40 (h,k) pairs
  const int wv = tid >> 5, ln = tid & 31;
  for (int p = wv; p < 2 * KN; p += 8) {
    int h = p / KN, k = p % KN;
    float s = 0.0f;
    for (int f = ln; f < KV_F; f += 32)
      s += (float)kv[k][f] * (float)g[h][f];
    #pragma unroll
    for (int off = 16; off > 0; off >>= 1) s += __shfl_xor(s, off, 32);
    if (ln == 0) sc[h][k] = s * 0.09901475f;   // 1/sqrt(102)
  }
  __syncthreads();

  // phase 3: masked softmax (2 heads, serial per thread)
  if (tid < 2) {
    float m = -1e30f, buf[KN];
    for (int k = 0; k < KN; ++k) {
      float x = sc[tid][k];
      if (sng[k] == 0) x = -1e9f;              // node 0 = padding
      buf[k] = x; m = fmaxf(m, x);
    }
    float sum = 0.0f;
    for (int k = 0; k < KN; ++k) { float e = __expf(buf[k] - m); at[tid][k] = e; sum += e; }
    float inv = 1.0f / sum;
    for (int k = 0; k < KN; ++k) at[tid][k] *= inv;
  }
  __syncthreads();

  // phase 4: ctx[h][f] = sum_k attn[h][k] * kv[k][f]
  for (int idx = tid; idx < 2 * (KV_LD / 4); idx += 256) {
    int h = idx / (KV_LD / 4);
    int f = (idx % (KV_LD / 4)) * 4;
    float a0 = 0, a1 = 0, a2 = 0, a3 = 0;
    for (int k = 0; k < KN; ++k) {
      float w = at[h][k];
      const _Float16* p = &kv[k][f];
      a0 += w * (float)p[0]; a1 += w * (float)p[1];
      a2 += w * (float)p[2]; a3 += w * (float)p[3];
    }
    _Float16* o = (h ? ctx1 : ctx0) + (size_t)q * KV_LD + f;
    o[0] = (_Float16)a0; o[1] = (_Float16)a1; o[2] = (_Float16)a2; o[3] = (_Float16)a3;
  }
}

// ---------------------------------------------------------------------------
extern "C" void kernel_launch(void* const* d_in, const int* in_sizes, int n_in,
                              void* d_out, int out_size, void* d_ws, size_t ws_size,
                              hipStream_t stream) {
  const float* nf  = (const float*)d_in[0];
  const float* ef  = (const float*)d_in[1];
  const float* mem = (const float*)d_in[2];
  const float* tw  = (const float*)d_in[3];
  const float* tb  = (const float*)d_in[4];
  const float* Wq  = (const float*)d_in[5];
  const float* Wk  = (const float*)d_in[6];
  const float* Wv  = (const float*)d_in[7];
  const float* Wm1 = (const float*)d_in[8];
  const float* bm1 = (const float*)d_in[9];
  const float* Wm2 = (const float*)d_in[10];
  const float* bm2 = (const float*)d_in[11];
  const float* ts  = (const float*)d_in[12];
  const float* ntm = (const float*)d_in[13];
  const int* nodes = (const int*)d_in[14];
  const int* nghs  = (const int*)d_in[15];
  const int* eidx  = (const int*)d_in[16];

  const int Q = in_sizes[12];          // 12288 (divisible by 16)
  const int mTiles = Q / 16;

  // ---- workspace carve-out (256B aligned) ----
  char* base = (char*)d_ws;
  size_t off = 0;
  auto alloc = [&](size_t bytes) -> void* {
    void* p = base + off;
    off = (off + bytes + 255) & ~(size_t)255;
    return p;
  };
  _Float16* BTq  = (_Float16*)alloc((size_t)208 * A1_LD * 2);
  _Float16* BT2a = (_Float16*)alloc((size_t)544 * 128 * 2);
  _Float16* BT2b = (_Float16*)alloc((size_t)544 * 128 * 2);
  _Float16* BT3a = (_Float16*)alloc((size_t)112 * KV_LD * 2);
  _Float16* BT3b = (_Float16*)alloc((size_t)112 * KV_LD * 2);
  _Float16* BT4  = (_Float16*)alloc((size_t)208 * MRG_LD * 2);
  _Float16* BT5  = (_Float16*)alloc((size_t)208 * HID_LD * 2);
  float*    qb   = (float*)alloc(256 * 4);
  _Float16* A1   = (_Float16*)alloc((size_t)Q * A1_LD * 2);
  _Float16* qbuf = (_Float16*)alloc((size_t)Q * Q_LD * 2);
  _Float16* G0   = (_Float16*)alloc((size_t)Q * KV_LD * 2);
  _Float16* G1   = (_Float16*)alloc((size_t)Q * KV_LD * 2);
  _Float16* ctx0 = (_Float16*)alloc((size_t)Q * KV_LD * 2);
  _Float16* ctx1 = (_Float16*)alloc((size_t)Q * KV_LD * 2);
  _Float16* att  = (_Float16*)alloc((size_t)Q * ATT_LD * 2);
  _Float16* mrg  = (_Float16*)alloc((size_t)Q * MRG_LD * 2);
  _Float16* hid  = (_Float16*)alloc((size_t)Q * HID_LD * 2);

  auto blk = [](int tot) { return (tot + 255) / 256; };

  // ---- weight prep (tiny; weights stay hot in L2) ----
  pack_t<<<blk(208 * A1_LD), 256, 0, stream>>>(BTq, A1_LD, Wq, DDIM, 0, 0, DDIM, FDIM, 208 * A1_LD);
  pack_d<<<blk(544 * 128), 256, 0, stream>>>(BT2a, 128, Wk, DDIM, 0, 0, KV_F, DH, 544 * 128);
  pack_d<<<blk(544 * 128), 256, 0, stream>>>(BT2b, 128, Wk, DDIM, 0, DH, KV_F, DH, 544 * 128);
  pack_t<<<blk(112 * KV_LD), 256, 0, stream>>>(BT3a, KV_LD, Wv, DDIM, 0, 0, DH, KV_F, 112 * KV_LD);
  pack_t<<<blk(112 * KV_LD), 256, 0, stream>>>(BT3b, KV_LD, Wv, DDIM, DH, 0, DH, KV_F, 112 * KV_LD);
  pack_t<<<blk(208 * MRG_LD), 256, 0, stream>>>(BT4, MRG_LD, Wm1, DDIM, 0, 0, DDIM, 376, 208 * MRG_LD);
  pack_t<<<blk(208 * HID_LD), 256, 0, stream>>>(BT5, HID_LD, Wm2, DDIM, 0, 0, DDIM, DDIM, 208 * HID_LD);
  qbias_kernel<<<1, 256, 0, stream>>>(qb, tb, Wq);

  // ---- q projection: q = (nf+mem)[nodes] @ Wq[:172] + qbias, head-split layout ----
  build_a1<<<blk(Q * FDIM), 256, 0, stream>>>(A1, nf, mem, nodes, Q * FDIM);
  wmma_gemm<<<dim3(mTiles, 4), 128, 0, stream>>>(A1, A1_LD, A1_LD, BTq, A1_LD, 13, qb,
                                                 qbuf, nullptr, Q_LD, DDIM, 0, 0, 1);

  // ---- G_h = q_h @ Wk_h^T  ([Q,102]@[102,516] per head) ----
  wmma_gemm<<<dim3(mTiles, 9), 128, 0, stream>>>(qbuf,       Q_LD, 128, BT2a, 128, 34, nullptr,
                                                 G0, nullptr, KV_LD, KV_F, 0, 0, 0);
  wmma_gemm<<<dim3(mTiles, 9), 128, 0, stream>>>(qbuf + 104, Q_LD, 128, BT2b, 128, 34, nullptr,
                                                 G1, nullptr, KV_LD, KV_F, 0, 0, 0);

  // ---- gather + scores + softmax + ctx (memory-bound heart of the op) ----
  attn_kernel<<<Q, 256, 0, stream>>>(nf, ef, mem, tw, tb, ts, ntm, nghs, eidx,
                                     G0, G1, ctx0, ctx1);

  // ---- att_out_h = ctx_h @ Wv_h  ([Q,516]@[516,102] per head) ----
  wmma_gemm<<<dim3(mTiles, 2), 128, 0, stream>>>(ctx0, KV_LD, KV_LD, BT3a, KV_LD, 7, nullptr,
                                                 att, nullptr, ATT_LD, DH, 0, 0, 0);
  wmma_gemm<<<dim3(mTiles, 2), 128, 0, stream>>>(ctx1, KV_LD, KV_LD, BT3b, KV_LD, 7, nullptr,
                                                 att, nullptr, ATT_LD, DH, DH, 0, 0);

  // ---- merge MLP ----
  build_merged<<<blk(Q * 376), 256, 0, stream>>>(mrg, att, nf, nodes, Q * 376);
  wmma_gemm<<<dim3(mTiles, 4), 128, 0, stream>>>(mrg, MRG_LD, MRG_LD, BT4, MRG_LD, 13, bm1,
                                                 hid, nullptr, HID_LD, DDIM, 0, 1, 0);
  wmma_gemm<<<dim3(mTiles, 4), 128, 0, stream>>>(hid, HID_LD, HID_LD, BT5, HID_LD, 13, bm2,
                                                 nullptr, (float*)d_out, DDIM, DDIM, 0, 0, 0);
}
